// ADUPredictor_67044439490770
// MI455X (gfx1250) — matrix operand order repaired and checked by reading, not compiled, not run on hardware
//
#include <hip/hip_runtime.h>
#include <hip/hip_bf16.h>
#include <cstdint>
#include <cstddef>

#define DEV __device__ __forceinline__

typedef __bf16 bf16_t;
typedef __attribute__((ext_vector_type(16))) __bf16 v16bf;
typedef __attribute__((ext_vector_type(8)))  float  v8f;
typedef __attribute__((ext_vector_type(4)))  unsigned int u32x4;
typedef __attribute__((ext_vector_type(8)))  int i32x8;
typedef __attribute__((ext_vector_type(4)))  int i32x4;

struct Frag { uint4 lo, hi; };

constexpr int BATCH  = 2;
constexpr int SEQ    = 2048;
constexpr int DIM    = 768;
constexpr int NH     = 12;
constexpr int DHEAD  = 64;
constexpr int WIN    = 256;
constexpr int FFD    = 3072;
constexpr int NLAYER = 12;
constexpr int NCHUNK = SEQ / WIN;   // 8
constexpr int NTOK   = BATCH * SEQ; // 4096
constexpr int KW     = 3 * WIN;     // 768 keys per window

#if defined(__has_builtin)
#if __has_builtin(__builtin_amdgcn_tensor_load_to_lds) && __has_builtin(__builtin_amdgcn_s_wait_tensorcnt)
#define HAVE_TDM 1
#endif
#endif
#ifndef HAVE_TDM
#define HAVE_TDM 0
#endif

// ---------------- helpers ----------------

DEV bf16_t f2b(float f) {
  unsigned u = __float_as_uint(f);
  unsigned r = (u + 0x7FFFu + ((u >> 16) & 1u)) >> 16;
  unsigned short s = (unsigned short)r;
  return __builtin_bit_cast(bf16_t, s);
}

DEV unsigned pack2(float a, float b) {
  unsigned short l = __builtin_bit_cast(unsigned short, f2b(a));
  unsigned short h = __builtin_bit_cast(unsigned short, f2b(b));
  return (unsigned)l | ((unsigned)h << 16);
}

DEV v16bf ld_frag(const bf16_t* p0, const bf16_t* p1) {
  Frag f;
  f.lo = *(const uint4*)p0;
  f.hi = *(const uint4*)p1;
  return __builtin_bit_cast(v16bf, f);
}

DEV v8f wmma_bf16(v16bf a, v16bf b, v8f c) {
  return __builtin_amdgcn_wmma_f32_16x16x32_bf16(false, a, false, b, (short)0, c, false, false);
}

DEV float sigmoidf_(float x) { return 1.0f / (1.0f + __expf(-x)); }

DEV float gelu_tanh(float v) {
  float v3 = v * v * v;
  return 0.5f * v * (1.0f + tanhf(0.7978845608f * (v + 0.044715f * v3)));
}

#if HAVE_TDM
// Tensor Data Mover: DMA a fully in-bounds 2D tile (tile_x elems of 2^dsz_log2
// bytes, tile_y rows, row stride stride_elems) from global to contiguous LDS.
// D# layout per CDNA5 ISA.
// This toolchain (clang-23 / therock-10.0) exposes the 6-argument builtin:
// (uint32x4 g0, int32x8 g1, int32x4 g2, int32x4 g3, int32x8 g4, i32 cpol).
DEV void tdm_load_2d(unsigned lds_byte_addr, const void* gptr,
                     unsigned tile_x, unsigned tile_y,
                     unsigned stride_elems, unsigned dsz_log2)
{
  unsigned long long ga = (unsigned long long)(uintptr_t)gptr;
  unsigned lds_u = (unsigned)__builtin_amdgcn_readfirstlane((int)lds_byte_addr);
  unsigned galo  = (unsigned)__builtin_amdgcn_readfirstlane((int)(unsigned)ga);
  unsigned gahi  = (unsigned)__builtin_amdgcn_readfirstlane((int)(unsigned)(ga >> 32));
  unsigned strd  = (unsigned)__builtin_amdgcn_readfirstlane((int)stride_elems);

  u32x4 g0;
  g0[0] = 1u;                                       // count=1 (user descriptor)
  g0[1] = lds_u;                                    // lds_addr [63:32]
  g0[2] = galo;                                     // global_addr [95:64]
  g0[3] = (gahi & 0x01FFFFFFu) | (2u << 30);        // global_addr[56:32] | type=2

  i32x8 g1;
  g1[0] = (int)(dsz_log2 << 16);                    // wg_mask=0 | data_size
  g1[1] = (int)((tile_x & 0xFFFFu) << 16);          // tensor_dim0[15:0] (== tile_x)
  g1[2] = (int)((tile_x >> 16) | ((tile_y & 0xFFFFu) << 16));   // td0 hi | td1 lo
  g1[3] = (int)((tile_y >> 16) | ((tile_x & 0xFFFFu) << 16));   // td1 hi | tile_dim0
  g1[4] = (int)(tile_y & 0xFFFFu);                  // tile_dim1 | tile_dim2=0
  g1[5] = (int)strd;                                // tensor_dim0_stride[31:0]
  g1[6] = 0;                                        // stride0 hi16 | stride1 lo16
  g1[7] = 0;                                        // stride1 hi32

  i32x4 z4;
  z4[0] = 0; z4[1] = 0; z4[2] = 0; z4[3] = 0;
  i32x8 z8;
  z8[0] = 0; z8[1] = 0; z8[2] = 0; z8[3] = 0;
  z8[4] = 0; z8[5] = 0; z8[6] = 0; z8[7] = 0;
  __builtin_amdgcn_tensor_load_to_lds(g0, g1, z4, z4, z8, 0);
}
#endif

// ---------------- generic WMMA GEMM ----------------
// C[M,N] = act( (A_bf16[M,K] @ B + bias + bias2) * alpha + beta*Cin )
// btrans==0: B is f32 [K][N] row-major; btrans==1: B is f32 [N][K] row-major.
// Requires M%128==0, N%64==0, K%32==0.

__global__ __launch_bounds__(256) void gemm_wmma_kernel(
    const bf16_t* __restrict__ A, const float* __restrict__ Bw,
    const float* __restrict__ bias, const float* __restrict__ bias2,
    const float* __restrict__ Cin, float beta,
    float* __restrict__ outF, bf16_t* __restrict__ outB,
    int M, int N, int K, int btrans, float alpha, int act)
{
  __shared__ __attribute__((aligned(16))) bf16_t sA[128 * 32];
  __shared__ __attribute__((aligned(16))) bf16_t sB[64 * 32];

  const int tid  = threadIdx.x;
  const int lane = tid & 31;
  const int wave = tid >> 5;
  const int wm   = wave & 3;   // 4 waves along M (32 rows each)
  const int wn   = wave >> 2;  // 2 waves along N (32 cols each)
  const int kh   = lane >> 4;
  const int ln15 = lane & 15;
  const int m0   = blockIdx.y * 128;
  const int n0   = blockIdx.x * 64;

  v8f acc[2][2];
  #pragma unroll
  for (int i = 0; i < 2; i++)
    #pragma unroll
    for (int j = 0; j < 2; j++)
      #pragma unroll
      for (int r = 0; r < 8; r++) acc[i][j][r] = 0.0f;

  for (int k0 = 0; k0 < K; k0 += 32) {
    __syncthreads();
#if HAVE_TDM
    // A tile [128][32] bf16 via Tensor Data Mover (one wave issues the DMA)
    if (wave == 0) {
      tdm_load_2d((unsigned)(uintptr_t)&sA[0],
                  &A[(size_t)m0 * K + k0],
                  /*tile_x=*/32, /*tile_y=*/128,
                  /*stride_elems=*/(unsigned)K, /*dsz_log2=*/1);
      __builtin_amdgcn_s_wait_tensorcnt(0);
    }
#else
    #pragma unroll
    for (int i = 0; i < 2; i++) {
      int idx = tid + i * 256;              // 512 uint4 total
      int r   = idx >> 2;
      int c8  = (idx & 3) * 8;
      *(uint4*)&sA[r * 32 + c8] =
          *(const uint4*)&A[(size_t)(m0 + r) * K + k0 + c8];
    }
#endif
    // stage B tile transposed [64 n][32 k] bf16 (f32 -> bf16, float4 vectorized)
    if (!btrans) {
      #pragma unroll
      for (int i = 0; i < 2; i++) {
        int idx = tid + i * 256;            // 512 float4 reads
        int kl  = idx >> 4;                 // 0..31
        int nl4 = (idx & 15) * 4;           // 0..60
        float4 f = *(const float4*)&Bw[(size_t)(k0 + kl) * N + n0 + nl4];
        sB[(nl4 + 0) * 32 + kl] = f2b(f.x);
        sB[(nl4 + 1) * 32 + kl] = f2b(f.y);
        sB[(nl4 + 2) * 32 + kl] = f2b(f.z);
        sB[(nl4 + 3) * 32 + kl] = f2b(f.w);
      }
    } else {
      #pragma unroll
      for (int i = 0; i < 2; i++) {
        int idx = tid + i * 256;
        int nl  = idx >> 3;                 // 0..63
        int kl4 = (idx & 7) * 4;            // 0..28
        float4 f = *(const float4*)&Bw[(size_t)(n0 + nl) * K + k0 + kl4];
        *(uint2*)&sB[nl * 32 + kl4] = make_uint2(pack2(f.x, f.y), pack2(f.z, f.w));
      }
    }
    __syncthreads();

    v16bf af[2], bfr[2];
    #pragma unroll
    for (int mt = 0; mt < 2; mt++) {
      const bf16_t* p = &sA[(wm * 32 + mt * 16 + ln15) * 32 + kh * 8];
      af[mt] = ld_frag(p, p + 16);
    }
    #pragma unroll
    for (int nt = 0; nt < 2; nt++) {
      const bf16_t* p = &sB[(wn * 32 + nt * 16 + ln15) * 32 + kh * 8];
      bfr[nt] = ld_frag(p, p + 16);
    }
    #pragma unroll
    for (int mt = 0; mt < 2; mt++)
      #pragma unroll
      for (int nt = 0; nt < 2; nt++)
        acc[mt][nt] = wmma_bf16(af[mt], bfr[nt], acc[mt][nt]);
  }

  // epilogue
  #pragma unroll
  for (int mt = 0; mt < 2; mt++)
    #pragma unroll
    for (int nt = 0; nt < 2; nt++) {
      int n  = n0 + wn * 32 + nt * 16 + ln15;
      int mb = m0 + wm * 32 + mt * 16 + kh * 8;
      float bv = 0.0f;
      if (bias)  bv += bias[n];
      if (bias2) bv += bias2[n];
      #pragma unroll
      for (int r = 0; r < 8; r++) {
        int m = mb + r;
        float v = (acc[mt][nt][r] + bv) * alpha;
        if (Cin) v += beta * Cin[(size_t)m * N + n];
        if (act == 1)      v = gelu_tanh(v);
        else if (act == 2) v = fmaxf(v, 0.0f);
        if (outF) outF[(size_t)m * N + n] = v;
        if (outB) outB[(size_t)m * N + n] = f2b(v);
      }
    }
}

// ---------------- embedding + LN ----------------

__global__ __launch_bounds__(256) void embed_ln_kernel(
    const int* __restrict__ sent, const float* __restrict__ wemb,
    const float* __restrict__ pemb, const float* __restrict__ gs,
    const float* __restrict__ gb, float* __restrict__ x, bf16_t* __restrict__ xb)
{
  __shared__ float red[256];
  int tok = blockIdx.x, tid = threadIdx.x;
  int sp  = tok & (SEQ - 1);
  int wid = sent[tok];
  float v[3];
  #pragma unroll
  for (int i = 0; i < 3; i++) {
    int j = tid + i * 256;
    v[i] = wemb[(size_t)wid * DIM + j] + pemb[(size_t)sp * DIM + j];
  }
  red[tid] = v[0] + v[1] + v[2];
  __syncthreads();
  for (int t = 128; t > 0; t >>= 1) { if (tid < t) red[tid] += red[tid + t]; __syncthreads(); }
  float mean = red[0] * (1.0f / DIM); __syncthreads();
  float d0 = v[0] - mean, d1 = v[1] - mean, d2 = v[2] - mean;
  red[tid] = d0 * d0 + d1 * d1 + d2 * d2;
  __syncthreads();
  for (int t = 128; t > 0; t >>= 1) { if (tid < t) red[tid] += red[tid + t]; __syncthreads(); }
  float rs = rsqrtf(red[0] * (1.0f / DIM) + 1e-5f); __syncthreads();
  #pragma unroll
  for (int i = 0; i < 3; i++) {
    int j = tid + i * 256;
    float o = (v[i] - mean) * rs * gs[j] + gb[j];
    x[(size_t)tok * DIM + j]  = o;
    xb[(size_t)tok * DIM + j] = f2b(o);
  }
}

// residual + LN:  x = LN(x + delta) ; also emit bf16 copy
__global__ __launch_bounds__(256) void resln_kernel(
    float* __restrict__ x, const float* __restrict__ delta,
    const float* __restrict__ gs, const float* __restrict__ gb,
    bf16_t* __restrict__ xb)
{
  __shared__ float red[256];
  int tok = blockIdx.x, tid = threadIdx.x;
  float v[3];
  #pragma unroll
  for (int i = 0; i < 3; i++) {
    int j = tid + i * 256;
    v[i] = x[(size_t)tok * DIM + j] + delta[(size_t)tok * DIM + j];
  }
  red[tid] = v[0] + v[1] + v[2];
  __syncthreads();
  for (int t = 128; t > 0; t >>= 1) { if (tid < t) red[tid] += red[tid + t]; __syncthreads(); }
  float mean = red[0] * (1.0f / DIM); __syncthreads();
  float d0 = v[0] - mean, d1 = v[1] - mean, d2 = v[2] - mean;
  red[tid] = d0 * d0 + d1 * d1 + d2 * d2;
  __syncthreads();
  for (int t = 128; t > 0; t >>= 1) { if (tid < t) red[tid] += red[tid + t]; __syncthreads(); }
  float rs = rsqrtf(red[0] * (1.0f / DIM) + 1e-5f); __syncthreads();
  #pragma unroll
  for (int i = 0; i < 3; i++) {
    int j = tid + i * 256;
    float o = (v[i] - mean) * rs * gs[j] + gb[j];
    x[(size_t)tok * DIM + j]  = o;
    xb[(size_t)tok * DIM + j] = f2b(o);
  }
}

// ---------------- sliding-window attention (WMMA) ----------------
// scores[h][c][qi][ki] for one batch element b; q already scaled by 1/8.

__global__ __launch_bounds__(256) void attn_scores_kernel(
    const bf16_t* __restrict__ qb, const bf16_t* __restrict__ kb,
    float* __restrict__ sc, int b)
{
  __shared__ __attribute__((aligned(16))) bf16_t sQ[WIN * DHEAD];   // 32 KB
  __shared__ __attribute__((aligned(16))) bf16_t sK[192 * DHEAD];   // 24 KB
  const int c = blockIdx.x, h = blockIdx.y;
  const int tid = threadIdx.x, lane = tid & 31, w = tid >> 5;
  const int kh = lane >> 4, ln15 = lane & 15;

  // stage all 256 q rows for this (b,h,c)
#if HAVE_TDM
  if (w == 0) {
    tdm_load_2d((unsigned)(uintptr_t)&sQ[0],
                &qb[(size_t)(b * SEQ + c * WIN) * DIM + h * DHEAD],
                /*tile_x=*/DHEAD, /*tile_y=*/WIN,
                /*stride_elems=*/DIM, /*dsz_log2=*/1);
    __builtin_amdgcn_s_wait_tensorcnt(0);
  }
#else
  #pragma unroll
  for (int i = 0; i < 8; i++) {
    int idx = tid + i * 256;              // 2048 uint4
    int r = idx >> 3, c8 = (idx & 7) * 8;
    *(uint4*)&sQ[r * DHEAD + c8] =
        *(const uint4*)&qb[(size_t)(b * SEQ + c * WIN + r) * DIM + h * DHEAD + c8];
  }
#endif
  __syncthreads();

  // per-wave A fragments (2 M-tiles x 2 K-steps), kept in registers
  v16bf af[2][2];
  #pragma unroll
  for (int j = 0; j < 2; j++) {
    int mrow = (2 * w + j) * 16 + ln15;
    #pragma unroll
    for (int ks = 0; ks < 2; ks++) {
      const bf16_t* p = &sQ[mrow * DHEAD + ks * 32 + kh * 8];
      af[j][ks] = ld_frag(p, p + 16);
    }
  }

  for (int ck = 0; ck < 4; ck++) {        // 4 chunks of 192 keys
    __syncthreads();
    #pragma unroll
    for (int i = 0; i < 6; i++) {
      int idx = tid + i * 256;            // 1536 uint4
      int kl = idx >> 3, c8 = (idx & 7) * 8;
      int sk = c * WIN - WIN + ck * 192 + kl;
      uint4 val;
      if (sk >= 0 && sk < SEQ)
        val = *(const uint4*)&kb[(size_t)(b * SEQ + sk) * DIM + h * DHEAD + c8];
      else
        val = make_uint4(0u, 0u, 0u, 0u);
      *(uint4*)&sK[kl * DHEAD + c8] = val;
    }
    __syncthreads();

    for (int nt = 0; nt < 12; nt++) {
      v16bf bfr[2];
      #pragma unroll
      for (int ks = 0; ks < 2; ks++) {
        const bf16_t* p = &sK[(nt * 16 + ln15) * DHEAD + ks * 32 + kh * 8];
        bfr[ks] = ld_frag(p, p + 16);
      }
      #pragma unroll
      for (int j = 0; j < 2; j++) {
        v8f acc;
        #pragma unroll
        for (int r = 0; r < 8; r++) acc[r] = 0.0f;
        acc = wmma_bf16(af[j][0], bfr[0], acc);
        acc = wmma_bf16(af[j][1], bfr[1], acc);

        int ki = ck * 192 + nt * 16 + ln15;
        int pk = c * WIN - WIN + ki;
        int qbase = (2 * w + j) * 16 + kh * 8;
        #pragma unroll
        for (int r = 0; r < 8; r++) {
          int qi  = qbase + r;
          int rel = ki - WIN - qi;
          bool valid = (rel >= -WIN) && (rel <= WIN) && (pk >= 0) && (pk < SEQ);
          float v = valid ? acc[r] : -1e9f;
          sc[((size_t)(h * NCHUNK + c) * WIN + qi) * KW + ki] = v;
        }
      }
    }
  }
}

__global__ __launch_bounds__(256) void softmax_win_kernel(float* __restrict__ sc)
{
  __shared__ float red[256];
  int row = blockIdx.x, tid = threadIdx.x;
  float* p = sc + (size_t)row * KW;
  float a0 = p[tid], a1 = p[tid + 256], a2 = p[tid + 512];
  red[tid] = fmaxf(a0, fmaxf(a1, a2));
  __syncthreads();
  for (int t = 128; t > 0; t >>= 1) { if (tid < t) red[tid] = fmaxf(red[tid], red[tid + t]); __syncthreads(); }
  float mx = red[0]; __syncthreads();
  float e0 = __expf(a0 - mx), e1 = __expf(a1 - mx), e2 = __expf(a2 - mx);
  red[tid] = e0 + e1 + e2;
  __syncthreads();
  for (int t = 128; t > 0; t >>= 1) { if (tid < t) red[tid] += red[tid + t]; __syncthreads(); }
  float inv = 1.0f / red[0]; __syncthreads();
  p[tid] = e0 * inv; p[tid + 256] = e1 * inv; p[tid + 512] = e2 * inv;
}

__global__ __launch_bounds__(256) void attn_ctx_kernel(
    const float* __restrict__ sc, const bf16_t* __restrict__ vb,
    bf16_t* __restrict__ ctxb, int b)
{
  __shared__ __attribute__((aligned(16))) bf16_t sV[DHEAD * 136]; // [d][128+pad]
  const int c = blockIdx.x, h = blockIdx.y;
  const int tid = threadIdx.x, lane = tid & 31, w = tid >> 5;
  const int kh = lane >> 4, ln15 = lane & 15;

  v8f acc[2][4];
  #pragma unroll
  for (int j = 0; j < 2; j++)
    #pragma unroll
    for (int nt = 0; nt < 4; nt++)
      #pragma unroll
      for (int r = 0; r < 8; r++) acc[j][nt][r] = 0.0f;

  for (int kc = 0; kc < 6; kc++) {        // 6 chunks of 128 keys
    __syncthreads();
    #pragma unroll
    for (int i = 0; i < 32; i++) {
      int idx = tid + i * 256;            // 8192 elems
      int dl = idx & 63, kl = idx >> 6;
      int sk = c * WIN - WIN + kc * 128 + kl;
      bf16_t v = f2b(0.0f);
      if (sk >= 0 && sk < SEQ)
        v = vb[(size_t)(b * SEQ + sk) * DIM + h * DHEAD + dl];
      sV[dl * 136 + kl] = v;
    }
    __syncthreads();

    for (int ks = 0; ks < 4; ks++) {      // K-steps of 32 within chunk
      v16bf af[2];
      #pragma unroll
      for (int j = 0; j < 2; j++) {
        int qi = (2 * w + j) * 16 + ln15;
        const float* p = sc + ((size_t)(h * NCHUNK + c) * WIN + qi) * KW
                            + kc * 128 + ks * 32 + kh * 8;
        float4 f0 = *(const float4*)p;
        float4 f1 = *(const float4*)(p + 4);
        float4 f2 = *(const float4*)(p + 16);
        float4 f3 = *(const float4*)(p + 20);
        Frag fr;
        fr.lo = make_uint4(pack2(f0.x, f0.y), pack2(f0.z, f0.w),
                           pack2(f1.x, f1.y), pack2(f1.z, f1.w));
        fr.hi = make_uint4(pack2(f2.x, f2.y), pack2(f2.z, f2.w),
                           pack2(f3.x, f3.y), pack2(f3.z, f3.w));
        af[j] = __builtin_bit_cast(v16bf, fr);
      }
      #pragma unroll
      for (int nt = 0; nt < 4; nt++) {
        const bf16_t* p = &sV[(nt * 16 + ln15) * 136 + ks * 32 + kh * 8];
        v16bf bfr = ld_frag(p, p + 16);
        #pragma unroll
        for (int j = 0; j < 2; j++)
          acc[j][nt] = wmma_bf16(af[j], bfr, acc[j][nt]);
      }
    }
  }

  #pragma unroll
  for (int j = 0; j < 2; j++)
    #pragma unroll
    for (int nt = 0; nt < 4; nt++) {
      int n = nt * 16 + ln15;
      int qbase = (2 * w + j) * 16 + kh * 8;
      #pragma unroll
      for (int r = 0; r < 8; r++) {
        int qi = qbase + r;
        ctxb[(size_t)(b * SEQ + c * WIN + qi) * DIM + h * DHEAD + n] =
            f2b(acc[j][nt][r]);
      }
    }
}

// ---------------- LSTM1 cell ----------------

__global__ __launch_bounds__(256) void lstm_cell_kernel(
    const float* __restrict__ z, float* __restrict__ cst, bf16_t* __restrict__ hB,
    float* __restrict__ l1o, bf16_t* __restrict__ l1ob,
    int dir, int t, int first)
{
  int row = blockIdx.x, u = threadIdx.x;
  const float* zr = z + (size_t)row * 1024;
  float zi = zr[u], zf = zr[256 + u], zg = zr[512 + u], zo = zr[768 + u];
  float cprev = first ? 0.0f : cst[(size_t)row * 256 + u];
  float cc = sigmoidf_(zf) * cprev + sigmoidf_(zi) * tanhf(zg);
  float hh = sigmoidf_(zo) * tanhf(cc);
  cst[(size_t)row * 256 + u] = cc;
  hB[(size_t)row * 256 + u]  = f2b(hh);
  size_t off = ((size_t)t * SEQ + row) * 512 + dir * 256 + u;
  l1o[off]  = hh;
  l1ob[off] = f2b(hh);
}

// ---------------- pooling head ----------------

__global__ __launch_bounds__(256) void energy_kernel(
    const float* __restrict__ e1, const float* __restrict__ w2,
    const float* __restrict__ b2, float* __restrict__ en)
{
  int row = blockIdx.x * 256 + threadIdx.x;
  const float* p = e1 + (size_t)row * 64;
  float a = b2[0];
  #pragma unroll
  for (int j = 0; j < 64; j++) a += p[j] * w2[j];
  en[row] = a;
}

__global__ __launch_bounds__(256) void softmax_seq_kernel(float* __restrict__ e)
{
  __shared__ float red[256];
  int b = blockIdx.x, tid = threadIdx.x;
  float* p = e + (size_t)b * SEQ;
  float vals[8];
  float mx = -3.4e38f;
  #pragma unroll
  for (int i = 0; i < 8; i++) { vals[i] = p[tid + i * 256]; mx = fmaxf(mx, vals[i]); }
  red[tid] = mx; __syncthreads();
  for (int t = 128; t > 0; t >>= 1) { if (tid < t) red[tid] = fmaxf(red[tid], red[tid + t]); __syncthreads(); }
  mx = red[0]; __syncthreads();
  float s = 0.0f;
  #pragma unroll
  for (int i = 0; i < 8; i++) { vals[i] = __expf(vals[i] - mx); s += vals[i]; }
  red[tid] = s; __syncthreads();
  for (int t = 128; t > 0; t >>= 1) { if (tid < t) red[tid] += red[tid + t]; __syncthreads(); }
  float inv = 1.0f / red[0]; __syncthreads();
  #pragma unroll
  for (int i = 0; i < 8; i++) p[tid + i * 256] = vals[i] * inv;
}

__global__ __launch_bounds__(256) void pool_kernel(
    const float* __restrict__ l1o, const float* __restrict__ wts,
    float* __restrict__ pooled)
{
  int b = blockIdx.x, tid = threadIdx.x;
  #pragma unroll
  for (int half = 0; half < 2; half++) {
    int j = tid + half * 256;
    float acc = 0.0f;
    for (int s = 0; s < SEQ; s++)
      acc += l1o[((size_t)b * SEQ + s) * 512 + j] * wts[b * SEQ + s];
    pooled[b * 512 + j] = acc;
  }
}

// ---------------- LSTM2 + tag (single block) ----------------

__global__ __launch_bounds__(256) void lstm2_tag_kernel(
    const float* __restrict__ pooled,
    const float* __restrict__ Wih, const float* __restrict__ Whh,
    const float* __restrict__ bih, const float* __restrict__ bhh,
    const float* __restrict__ Wtag, const float* __restrict__ btag,
    float* __restrict__ out)
{
  __shared__ float sx[512];
  __shared__ float sh[256];
  __shared__ float hsave[2][512];
  __shared__ float red[256];
  int tid = threadIdx.x;

  for (int d = 0; d < 2; d++) {
    float cc = 0.0f;
    sh[tid] = 0.0f;
    __syncthreads();
    for (int s = 0; s < 2; s++) {
      int t = (d == 0) ? s : 1 - s;
      sx[tid]       = pooled[t * 512 + tid];
      sx[tid + 256] = pooled[t * 512 + 256 + tid];
      __syncthreads();
      float zg4[4];
      #pragma unroll
      for (int g = 0; g < 4; g++) {
        int row = d * 1024 + g * 256 + tid;
        float zz = bih[row] + bhh[row];
        const float* wr = Wih + (size_t)row * 512;
        for (int k = 0; k < 512; k++) zz += wr[k] * sx[k];
        const float* hr = Whh + (size_t)row * 256;
        for (int k = 0; k < 256; k++) zz += hr[k] * sh[k];
        zg4[g] = zz;
      }
      cc = sigmoidf_(zg4[1]) * cc + sigmoidf_(zg4[0]) * tanhf(zg4[2]);
      float hh = sigmoidf_(zg4[3]) * tanhf(cc);
      __syncthreads();
      sh[tid] = hh;
      hsave[t][d * 256 + tid] = hh;
      __syncthreads();
    }
  }

  for (int b = 0; b < 2; b++) {
    float p = hsave[b][tid] * Wtag[tid] + hsave[b][256 + tid] * Wtag[256 + tid];
    red[tid] = p; __syncthreads();
    for (int t = 128; t > 0; t >>= 1) { if (tid < t) red[tid] += red[tid + t]; __syncthreads(); }
    if (tid == 0) out[b] = red[0] + btag[0];
    __syncthreads();
  }
}

// ---------------- host orchestration ----------------

static void launch_gemm(hipStream_t st, const bf16_t* A, const float* Bw,
                        const float* bias, const float* bias2,
                        const float* Cin, float beta,
                        float* outF, bf16_t* outB,
                        int M, int N, int K, int btrans, float alpha, int act)
{
  dim3 grid(N / 64, M / 128);
  gemm_wmma_kernel<<<grid, 256, 0, st>>>(A, Bw, bias, bias2, Cin, beta,
                                         outF, outB, M, N, K, btrans, alpha, act);
}

extern "C" void kernel_launch(void* const* d_in, const int* in_sizes, int n_in,
                              void* d_out, int out_size, void* d_ws, size_t ws_size,
                              hipStream_t stream)
{
  (void)in_sizes; (void)n_in; (void)out_size; (void)ws_size;

  const int*   sent    = (const int*)  d_in[0];
  const float* word_e  = (const float*)d_in[1];
  const float* pos_e   = (const float*)d_in[2];
  const float* eln_s   = (const float*)d_in[3];
  const float* eln_b   = (const float*)d_in[4];
  const float* Wq      = (const float*)d_in[5];
  const float* Wk      = (const float*)d_in[6];
  const float* Wv      = (const float*)d_in[7];
  const float* Wo      = (const float*)d_in[8];
  const float* bq      = (const float*)d_in[9];
  const float* bk      = (const float*)d_in[10];
  const float* bv      = (const float*)d_in[11];
  const float* bo      = (const float*)d_in[12];
  const float* ln1_s   = (const float*)d_in[13];
  const float* ln1_b   = (const float*)d_in[14];
  const float* Wf1     = (const float*)d_in[15];
  const float* bf1     = (const float*)d_in[16];
  const float* Wf2     = (const float*)d_in[17];
  const float* bf2     = (const float*)d_in[18];
  const float* ln2_s   = (const float*)d_in[19];
  const float* ln2_b   = (const float*)d_in[20];
  const float* l1_Wih  = (const float*)d_in[21];
  const float* l1_Whh  = (const float*)d_in[22];
  const float* l1_bih  = (const float*)d_in[23];
  const float* l1_bhh  = (const float*)d_in[24];
  const float* pa_W1   = (const float*)d_in[25];
  const float* pa_b1   = (const float*)d_in[26];
  const float* pa_W2   = (const float*)d_in[27];
  const float* pa_b2   = (const float*)d_in[28];
  const float* l2_Wih  = (const float*)d_in[29];
  const float* l2_Whh  = (const float*)d_in[30];
  const float* l2_bih  = (const float*)d_in[31];
  const float* l2_bhh  = (const float*)d_in[32];
  const float* Wtag    = (const float*)d_in[33];
  const float* btag    = (const float*)d_in[34];
  float* outp = (float*)d_out;

  // workspace carve-out
  char* wsb = (char*)d_ws;
  size_t off = 0;
  auto carve = [&](size_t bytes) -> void* {
    void* p = wsb + off;
    off = (off + bytes + 255) & ~(size_t)255;
    return p;
  };
  float*  x     = (float*) carve((size_t)NTOK * DIM * 4);
  bf16_t* xb    = (bf16_t*)carve((size_t)NTOK * DIM * 2);
  bf16_t* qbuf  = (bf16_t*)carve((size_t)NTOK * DIM * 2);
  bf16_t* kbuf  = (bf16_t*)carve((size_t)NTOK * DIM * 2);
  bf16_t* vbuf  = (bf16_t*)carve((size_t)NTOK * DIM * 2);
  float*  sc    = (float*) carve((size_t)NH * NCHUNK * WIN * KW * 4); // 75.5 MB, reused per b
  bf16_t* ctxb  = (bf16_t*)carve((size_t)NTOK * DIM * 2);
  float*  tmp   = (float*) carve((size_t)NTOK * DIM * 4);
  bf16_t* hff   = (bf16_t*)carve((size_t)NTOK * FFD * 2);
  float*  zbuf  = (float*) carve((size_t)SEQ * 1024 * 4);
  float*  cst   = (float*) carve((size_t)SEQ * 256 * 4);
  bf16_t* hB    = (bf16_t*)carve((size_t)SEQ * 256 * 2);
  float*  l1o   = (float*) carve((size_t)2 * SEQ * 512 * 4);
  bf16_t* l1ob  = (bf16_t*)carve((size_t)2 * SEQ * 512 * 2);
  float*  e1    = (float*) carve((size_t)NTOK * 64 * 4);
  float*  en    = (float*) carve((size_t)NTOK * 4);
  float*  pooled= (float*) carve((size_t)2 * 512 * 4);

  // 1) embedding + LN
  embed_ln_kernel<<<NTOK, 256, 0, stream>>>(sent, word_e, pos_e, eln_s, eln_b, x, xb);

  // 2) transformer layers
  for (int l = 0; l < NLAYER; l++) {
    const float* Wq_l = Wq + (size_t)l * DIM * DIM;
    const float* Wk_l = Wk + (size_t)l * DIM * DIM;
    const float* Wv_l = Wv + (size_t)l * DIM * DIM;
    const float* Wo_l = Wo + (size_t)l * DIM * DIM;
    const float* bq_l = bq + (size_t)l * DIM;
    const float* bk_l = bk + (size_t)l * DIM;
    const float* bv_l = bv + (size_t)l * DIM;
    const float* bo_l = bo + (size_t)l * DIM;
    const float* Wf1_l = Wf1 + (size_t)l * DIM * FFD;
    const float* bf1_l = bf1 + (size_t)l * FFD;
    const float* Wf2_l = Wf2 + (size_t)l * FFD * DIM;
    const float* bf2_l = bf2 + (size_t)l * DIM;

    // q (scaled by 1/sqrt(64)), k, v
    launch_gemm(stream, xb, Wq_l, bq_l, nullptr, nullptr, 0.f, nullptr, qbuf,
                NTOK, DIM, DIM, 0, 0.125f, 0);
    launch_gemm(stream, xb, Wk_l, bk_l, nullptr, nullptr, 0.f, nullptr, kbuf,
                NTOK, DIM, DIM, 0, 1.0f, 0);
    launch_gemm(stream, xb, Wv_l, bv_l, nullptr, nullptr, 0.f, nullptr, vbuf,
                NTOK, DIM, DIM, 0, 1.0f, 0);

    for (int b = 0; b < BATCH; b++) {
      dim3 ag(NCHUNK, NH);
      attn_scores_kernel<<<ag, 256, 0, stream>>>(qbuf, kbuf, sc, b);
      softmax_win_kernel<<<NH * NCHUNK * WIN, 256, 0, stream>>>(sc);
      attn_ctx_kernel<<<ag, 256, 0, stream>>>(sc, vbuf, ctxb, b);
    }

    // output proj + residual + LN1
    launch_gemm(stream, ctxb, Wo_l, bo_l, nullptr, nullptr, 0.f, tmp, nullptr,
                NTOK, DIM, DIM, 0, 1.0f, 0);
    resln_kernel<<<NTOK, 256, 0, stream>>>(x, tmp,
        ln1_s + (size_t)l * DIM, ln1_b + (size_t)l * DIM, xb);

    // FFN
    launch_gemm(stream, xb, Wf1_l, bf1_l, nullptr, nullptr, 0.f, nullptr, hff,
                NTOK, FFD, DIM, 0, 1.0f, 1 /*gelu*/);
    launch_gemm(stream, hff, Wf2_l, bf2_l, nullptr, nullptr, 0.f, tmp, nullptr,
                NTOK, DIM, FFD, 0, 1.0f, 0);
    resln_kernel<<<NTOK, 256, 0, stream>>>(x, tmp,
        ln2_s + (size_t)l * DIM, ln2_b + (size_t)l * DIM, xb);
  }

  // 3) BiLSTM-1 (scan over B axis: 2 steps, batch = SEQ)
  for (int d = 0; d < 2; d++) {
    for (int s = 0; s < 2; s++) {
      int t = (d == 0) ? s : 1 - s;
      launch_gemm(stream, xb + (size_t)t * SEQ * DIM,
                  l1_Wih + (size_t)d * 1024 * DIM,
                  l1_bih + (size_t)d * 1024, l1_bhh + (size_t)d * 1024,
                  nullptr, 0.f, zbuf, nullptr,
                  SEQ, 1024, DIM, 1 /*trans*/, 1.0f, 0);
      if (s == 1)
        launch_gemm(stream, hB, l1_Whh + (size_t)d * 1024 * 256,
                    nullptr, nullptr, zbuf, 1.0f, zbuf, nullptr,
                    SEQ, 1024, 256, 1, 1.0f, 0);
      lstm_cell_kernel<<<SEQ, 256, 0, stream>>>(zbuf, cst, hB, l1o, l1ob,
                                                d, t, (s == 0) ? 1 : 0);
    }
  }

  // 4) attention pooling head
  launch_gemm(stream, l1ob, pa_W1, pa_b1, nullptr, nullptr, 0.f, e1, nullptr,
              NTOK, 64, 512, 0, 1.0f, 2 /*relu*/);
  energy_kernel<<<NTOK / 256, 256, 0, stream>>>(e1, pa_W2, pa_b2, en);
  softmax_seq_kernel<<<BATCH, 256, 0, stream>>>(en);
  pool_kernel<<<BATCH, 256, 0, stream>>>(l1o, en, pooled);

  // 5) BiLSTM-2 + tag
  lstm2_tag_kernel<<<1, 256, 0, stream>>>(pooled, l2_Wih, l2_Whh, l2_bih, l2_bhh,
                                          Wtag, btag, outp);
}
